// ChannelSelection_39152921870889
// MI455X (gfx1250) — compile-verified, matrix operand order checked by reading, not run on hardware
//
#include <hip/hip_runtime.h>

// ---------------------------------------------------------------------------
// ChannelSelection for MI455X (gfx1250, wave32)
//   N=4, C=256, H=W=256, P=32  -> nh=nw=8, 64 patches, k=128 kept channels
// Pass 1: score + per-patch per-channel sums  (268 MB read)
// Pass 2: per-patch channel ranking via v_wmma_f32_16x16x32_f16 (tiny)
// Pass 3: masked copy (268 MB read + 268 MB write)
// ---------------------------------------------------------------------------

typedef __attribute__((ext_vector_type(16))) _Float16 v16h;
typedef __attribute__((ext_vector_type(8)))  float    v8f;

#define CS_N   4
#define CS_C   256
#define CS_H   256
#define CS_W   256
#define CS_P   32
#define CS_NH  8
#define CS_NP  64
#define CS_K   128

// ---------------------------------------------------------------------------
// Kernel A: fused score (0.5*(|dx|+|dy|) + 0.5*|x|) + 32x32 patch sums.
// grid = N*C*nh blocks, 256 threads. Wave w owns patch column w (32 lanes =
// 32 columns of one patch). One streaming load per pixel; horizontal
// neighbors via wave32 shuffles, vertical via register rotation.
// ---------------------------------------------------------------------------
__global__ __launch_bounds__(256) void cs_score_patchsum(
    const float* __restrict__ x, float* __restrict__ sp) {
  const int b  = blockIdx.x;
  const int ph = b & 7;
  const int c  = (b >> 3) & 255;
  const int n  = b >> 11;
  const int t  = threadIdx.x;
  const int w  = t;
  const int lane = t & 31;

  const float* plane = x + ((size_t)(n * CS_C + c)) * (CS_H * CS_W);
  const int h0 = ph * CS_P;

  float cur  = plane[h0 * CS_W + w];
  float prev = (h0 > 0) ? plane[(h0 - 1) * CS_W + w] : 0.0f;
  float acc  = 0.0f;

#pragma unroll 4
  for (int r = 0; r < CS_P; ++r) {
    const int h = h0 + r;
    float next = (h < CS_H - 1) ? plane[(h + 1) * CS_W + w] : 0.0f;

    float left = __shfl_up(cur, 1, 32);
    if (lane == 0)  left  = (w > 0) ? plane[h * CS_W + w - 1] : 0.0f;
    float right = __shfl_down(cur, 1, 32);
    if (lane == 31) right = (w < CS_W - 1) ? plane[h * CS_W + w + 1] : 0.0f;

    acc += 0.5f * (fabsf(right - left) + fabsf(next - prev)) + 0.5f * fabsf(cur);
    prev = cur;
    cur  = next;
  }

  // wave32 butterfly reduction -> patch sum (mean scale dropped: rank-invariant)
#pragma unroll
  for (int off = 16; off > 0; off >>= 1) acc += __shfl_xor(acc, off, 32);

  if (lane == 0) {
    const int pw = t >> 5;  // wave index == patch column
    sp[((n * CS_C + c) * CS_NP) + ph * 8 + pw] = acc;
  }
}

// ---------------------------------------------------------------------------
// Kernel B: per-(n,patch) channel ranking with WMMA.
//   rank[m] = #{k : sp[k] > sp[m]} + #{k < m : sp[k] == sp[m]}
// Built as G(16x32, f16 0/1) x ones(32x16) accumulated over 8 K-tiles with
// v_wmma_f32_16x16x32_f16. Exact (0/1 in f16, f32 accumulate).
// grid = N*64 blocks, 256 threads (8 waves x 2 M-tiles of 16 channels).
// sp staged to LDS via CDNA5 async global->LDS copy.
// ---------------------------------------------------------------------------
__global__ __launch_bounds__(256) void cs_rank_mask(
    const float* __restrict__ sp, float* __restrict__ mask) {
  __shared__ float s_sp[CS_C];

  const int bid = blockIdx.x;
  const int n = bid >> 6;
  const int p = bid & 63;
  const int t = threadIdx.x;

  // async global->LDS stage of sp[n, :, p] (one element per lane)
  {
    unsigned lds_addr =
        (unsigned)(unsigned long long)(__attribute__((address_space(3))) float*)&s_sp[t];
    unsigned long long gaddr =
        (unsigned long long)(sp + ((size_t)(n * CS_C + t)) * CS_NP + p);
    asm volatile("global_load_async_to_lds_b32 %0, %1, off"
                 :: "v"(lds_addr), "v"(gaddr) : "memory");
    asm volatile("s_wait_asynccnt 0" ::: "memory");
  }
  __syncthreads();

  const int wave = t >> 5;
  const int lane = t & 31;
  const int hi   = lane >> 4;   // K-half selector
  const int ml   = lane & 15;   // row within M-tile

  v16h bo;
#pragma unroll
  for (int i = 0; i < 16; ++i) bo[i] = (_Float16)1.0f;

  // each wave handles M-tiles {wave, wave+8} -> uniform control flow per wave
  for (int mt = wave; mt < 16; mt += 8) {
    const int   m_global = mt * 16 + ml;
    const float spm      = s_sp[m_global];

    v8f acc = {};
#pragma unroll
    for (int kt = 0; kt < 8; ++kt) {
      v16h a;
#pragma unroll
      for (int i = 0; i < 8; ++i) {
        // 16-bit A-matrix 16x32 layout: VGPR i<4 -> K = hi*8 + 2i(+1)
        //                               VGPR i>=4 -> K = 16 + hi*8 + 2(i-4)(+1)
        const int kk = (i < 4) ? (hi * 8 + 2 * i) : (16 + hi * 8 + 2 * (i - 4));
        const int k0 = kt * 32 + kk;
        const int k1 = k0 + 1;
        const float s0 = s_sp[k0];
        const float s1 = s_sp[k1];
        a[2 * i]     = (_Float16)(((s0 > spm) || (s0 == spm && k0 < m_global)) ? 1.0f : 0.0f);
        a[2 * i + 1] = (_Float16)(((s1 > spm) || (s1 == spm && k1 < m_global)) ? 1.0f : 0.0f);
      }
      acc = __builtin_amdgcn_wmma_f32_16x16x32_f16(
          /*neg_a=*/false, a, /*neg_b=*/false, bo,
          /*c_mod=*/(short)0, acc, /*reuse_a=*/false, /*reuse_b=*/false);
    }

    // D layout: lane 0 holds M=0..7 (VGPR0..7), lane 16 holds M=8..15
    if (ml == 0) {
#pragma unroll
      for (int r = 0; r < 8; ++r) {
        const int ch = mt * 16 + hi * 8 + r;
        mask[((n * CS_C + ch) << 6) + p] = (acc[r] < (float)CS_K) ? 1.0f : 0.0f;
      }
    }
  }
}

// ---------------------------------------------------------------------------
// Kernel C: out = x * mask (broadcast patch mask). float4-vectorized stream;
// a float4 never crosses a patch boundary (32 % 4 == 0). mask is L2-resident.
// ---------------------------------------------------------------------------
__global__ __launch_bounds__(256) void cs_apply_mask(
    const float4* __restrict__ x4, const float* __restrict__ mask,
    float4* __restrict__ out4) {
  const int idx = blockIdx.x * 256 + threadIdx.x;  // 16,777,216 float4s
  const int wq = idx & 63;           // float4 index within row (W/4 = 64)
  const int h  = (idx >> 6) & 255;
  const int c  = (idx >> 14) & 255;
  const int n  = idx >> 22;
  const int p  = ((h >> 5) << 3) + (wq >> 3);  // (h/32)*8 + w/32

  const float m = mask[(((n << 8) + c) << 6) + p];
  float4 v = x4[idx];
  v.x *= m; v.y *= m; v.z *= m; v.w *= m;
  out4[idx] = v;
}

// ---------------------------------------------------------------------------
extern "C" void kernel_launch(void* const* d_in, const int* in_sizes, int n_in,
                              void* d_out, int out_size, void* d_ws, size_t ws_size,
                              hipStream_t stream) {
  const float* x = (const float*)d_in[0];
  float* out = (float*)d_out;

  float* sp   = (float*)d_ws;                    // [N][C][64]  = 65536 f
  float* mask = sp + (CS_N * CS_C * CS_NP);      // [N][C][64]  = 65536 f

  // Pass 1: scores + patch sums
  cs_score_patchsum<<<CS_N * CS_C * CS_NH, 256, 0, stream>>>(x, sp);
  // Pass 2: top-k ranking via WMMA
  cs_rank_mask<<<CS_N * CS_NP, 256, 0, stream>>>(sp, mask);
  // Pass 3: masked copy
  const int n4 = (CS_N * CS_C * CS_H * CS_W) / 4;
  cs_apply_mask<<<n4 / 256, 256, 0, stream>>>((const float4*)x, mask, (float4*)out);
}
